// MHA_60868276519557
// MI455X (gfx1250) — compile-verified, matrix operand order checked by reading, not run on hardware
//
#include <hip/hip_runtime.h>
#include <hip/hip_bf16.h>
#include <stdint.h>

// ---------------------------------------------------------------------------
// MHA (GQA + RoPE) for MI455X / gfx1250.
//   - All matrix math on V_WMMA_F32_16X16X32_BF16 (bf16 in, f32 accumulate).
//   - GEMMs stream fragments through the 192MB L2 with global_prefetch_b8.
//   - Attention stages K/V tiles into LDS with the Tensor Data Mover
//     (tensor_load_to_lds + s_wait_tensorcnt), double-buffered.
//   - Softmax row reductions via ds_swizzle_b32 xor-mode (wave32 halves).
// ---------------------------------------------------------------------------

typedef __attribute__((ext_vector_type(16))) __bf16 bf16x16;
typedef __attribute__((ext_vector_type(8)))  __bf16 bf16x8;
typedef __attribute__((ext_vector_type(8)))  float  f32x8;

#define DEV __device__ __forceinline__

#if __has_builtin(__builtin_amdgcn_tensor_load_to_lds)
#define MHA_HAVE_TDM 1
#else
#define MHA_HAVE_TDM 0
#endif

DEV __bf16 f2bf(float f) {
  unsigned u = __builtin_bit_cast(unsigned, f);
  u += 0x7fffu + ((u >> 16) & 1u);              // round-to-nearest-even
  unsigned short s = (unsigned short)(u >> 16);
  return __builtin_bit_cast(__bf16, s);
}
DEV float bf2f(__bf16 b) {
  unsigned short s = __builtin_bit_cast(unsigned short, b);
  unsigned u = ((unsigned)s) << 16;
  return __builtin_bit_cast(float, u);
}

// ds_swizzle_b32 xor-mode within group-of-32: offset = {xor[14:10], or=0, and=0x1f}
#define SWZ_XOR(v, m) __builtin_bit_cast(float, \
    __builtin_amdgcn_ds_swizzle(__builtin_bit_cast(int, (v)), (0x1f | ((m) << 10))))

// Load a 16x32 bf16 fragment (A layout; also B when source is N x K).
// CDNA5 ISA 7.12.2: lanes 0-15 row=lane, K {0..7,16..23};
//                   lanes 16-31 row=lane-16, K {8..15,24..31}.
DEV bf16x16 load_frag(const __bf16* p, int ld) {
  const int lane = threadIdx.x & 31;
  const int r    = lane & 15;
  const int h    = lane >> 4;
  const __bf16* row = p + (size_t)r * ld;
  bf16x8 c0 = *reinterpret_cast<const bf16x8*>(row + 8 * h);
  bf16x8 c1 = *reinterpret_cast<const bf16x8*>(row + 16 + 8 * h);
  bf16x16 f;
#pragma unroll
  for (int i = 0; i < 8; ++i) { f[i] = c0[i]; f[8 + i] = c1[i]; }
  return f;
}

DEV f32x8 wmma_bf16(bf16x16 a, bf16x16 b, f32x8 c) {
  return __builtin_amdgcn_wmma_f32_16x16x32_bf16(
      false, a, false, b, (short)0, c, false, false);
}

DEV f32x8 zero8() { f32x8 z = {0.f,0.f,0.f,0.f,0.f,0.f,0.f,0.f}; return z; }

// ---------------------------------------------------------------------------
// Tensor Data Mover: 2D tile (bf16) global -> LDS.  D# per ISA ch.8:
// group0: count=1 | lds_addr | global_addr(57b) | type=2
// group1: data_size=1(2B); tensor_dim0=tile_dim0=tx; tensor_dim1=tile_dim1=ty;
//         tensor_dim0_stride = row pitch (elements).
// ---------------------------------------------------------------------------
#if MHA_HAVE_TDM
typedef __attribute__((ext_vector_type(4))) unsigned int u32x4;
typedef __attribute__((ext_vector_type(8))) int i32x8;
typedef __attribute__((ext_vector_type(4))) int i32x4;

DEV void tdm_load_2d(unsigned lds_addr, const void* gptr,
                     unsigned tx, unsigned ty, unsigned stride) {
  unsigned long long ga = (unsigned long long)(uintptr_t)gptr;
  u32x4 g0;
  g0[0] = 1u;                                             // count=1 (valid)
  g0[1] = lds_addr;                                       // LDS byte address
  g0[2] = (unsigned)ga;                                   // global addr lo
  g0[3] = (unsigned)((ga >> 32) & 0x01FFFFFFu) | (2u << 30); // hi | type=2
  i32x8 g1;
  g1[0] = (int)(1u << 16);                                // data_size=1 (2B)
  g1[1] = (int)((tx & 0xFFFFu) << 16);                    // tensor_dim0 lo16
  g1[2] = (int)(((tx >> 16) & 0xFFFFu) | ((ty & 0xFFFFu) << 16));
  g1[3] = (int)(((ty >> 16) & 0xFFFFu) | ((tx & 0xFFFFu) << 16)); // tile_dim0
  g1[4] = (int)(ty & 0xFFFFu);                            // tile_dim1
  g1[5] = (int)stride;                                    // dim0 stride lo32
  g1[6] = 0;
  g1[7] = 0;
  i32x4 gz = {0, 0, 0, 0};
#if defined(__clang_major__) && __clang_major__ >= 23
  i32x8 gz8 = {0, 0, 0, 0, 0, 0, 0, 0};
  __builtin_amdgcn_tensor_load_to_lds(g0, g1, gz, gz, gz8, 0);
#else
  __builtin_amdgcn_tensor_load_to_lds(g0, g1, gz, gz, 0);
#endif
}

DEV unsigned lds_off(const void* p) {
  // generic pointer to a __shared__ object: low 32 bits are the LDS address
  return (unsigned)(uintptr_t)p;
}
#endif

// ---------------------------------------------------------------------------
// fp32 -> bf16 converters
// ---------------------------------------------------------------------------
__global__ void cvt_bf16(const float* __restrict__ in, __bf16* __restrict__ out, int n) {
  int id = blockIdx.x * blockDim.x + threadIdx.x;
  if (id < n) out[id] = f2bf(in[id]);
}

// in: K x N row-major (fp32) -> out: N x K row-major (bf16)
__global__ void cvtT_bf16(const float* __restrict__ in, __bf16* __restrict__ out,
                          int K, int N) {
  int id = blockIdx.x * blockDim.x + threadIdx.x;
  if (id >= K * N) return;
  int n = id / K, k = id % K;
  out[(size_t)n * K + k] = f2bf(in[(size_t)k * N + n]);
}

// ---------------------------------------------------------------------------
// bf16 WMMA GEMM: C[M,N] = A[M,K] * (BT[N,K])^T.  8 waves/block, tile 128x64.
// Fragments streamed from L2; global_prefetch_b8 two K-steps ahead.
// ---------------------------------------------------------------------------
template <bool OUT_BF16>
__global__ __launch_bounds__(256) void gemm_wmma(const __bf16* __restrict__ A,
                                                 const __bf16* __restrict__ BT,
                                                 void* __restrict__ Cv,
                                                 int M, int N, int K) {
  const int lane = threadIdx.x & 31;
  const int wave = threadIdx.x >> 5;
  const int m0 = blockIdx.y * 128 + wave * 16;
  const int n0 = blockIdx.x * 64;

  f32x8 acc[4];
#pragma unroll
  for (int t = 0; t < 4; ++t) acc[t] = zero8();

  for (int k = 0; k < K; k += 32) {
    if (k + 64 < K) {
      __builtin_prefetch(A + (size_t)(m0 + (lane & 15)) * K + k + 64, 0, 3);
      __builtin_prefetch(BT + (size_t)(n0 + (lane & 15) * 4) * K + k + 64, 0, 3);
    }
    bf16x16 a = load_frag(A + (size_t)m0 * K + k, K);
#pragma unroll
    for (int t = 0; t < 4; ++t) {
      bf16x16 b = load_frag(BT + (size_t)(n0 + t * 16) * K + k, K);
      acc[t] = wmma_bf16(a, b, acc[t]);
    }
  }

  const int half = lane >> 4, nn = lane & 15;
#pragma unroll
  for (int t = 0; t < 4; ++t) {
#pragma unroll
    for (int j = 0; j < 8; ++j) {
      int m = m0 + j + 8 * half;
      int n = n0 + t * 16 + nn;
      size_t idx = (size_t)m * N + n;
      if (OUT_BF16) ((__bf16*)Cv)[idx] = f2bf(acc[t][j]);
      else          ((float*)Cv)[idx]  = acc[t][j];
    }
  }
}

// ---------------------------------------------------------------------------
// RoPE + head pack:  in [B,S,Hn,hd] -> out [B,Hn,S,hd] (roped)
// ---------------------------------------------------------------------------
__global__ void rope_pack(const __bf16* __restrict__ in, __bf16* __restrict__ out,
                          int Bn, int S, int Hn, int hd) {
  const int pairs = hd / 2;
  int id = blockIdx.x * blockDim.x + threadIdx.x;
  if (id >= Bn * S * Hn * pairs) return;
  int p  = id % pairs;
  int hh = (id / pairs) % Hn;
  int s  = (id / (pairs * Hn)) % S;
  int b  =  id / (pairs * Hn * S);

  size_t iin = (((size_t)(b * S + s) * Hn + hh) * hd) + 2 * p;
  float x1 = bf2f(in[iin]), x2 = bf2f(in[iin + 1]);
  float inv = __expf(-(float)p * (2.0f / (float)hd) * logf(10000.0f));
  float ang = (float)s * inv;
  float cs = cosf(ang), sn = sinf(ang);
  size_t io = (((size_t)(b * Hn + hh) * S + s) * hd) + 2 * p;
  out[io]     = f2bf(x1 * cs - x2 * sn);
  out[io + 1] = f2bf(x1 * sn + x2 * cs);
}

// V pack transposed: in [B,S,KV,hd] -> out [B,KV,hd,S]
__global__ void pack_vT(const __bf16* __restrict__ in, __bf16* __restrict__ out,
                        int Bn, int S, int KVn, int hd) {
  int id = blockIdx.x * blockDim.x + threadIdx.x;
  if (id >= Bn * S * KVn * hd) return;
  int d  = id % hd;
  int kv = (id / hd) % KVn;
  int s  = (id / (hd * KVn)) % S;
  int b  =  id / (hd * KVn * S);
  out[(((size_t)(b * KVn + kv) * hd + d) * S) + s] = in[id];
}

// ---------------------------------------------------------------------------
// Flash attention: one wave / 16 query rows / (b,h).  hd=128.
// K chunk (32 x 128) and V chunk (128 x 32) TDM'd into LDS, double buffered.
// S-tile 16x32 = 8 WMMAs; online softmax (ds_swizzle reductions); P through
// LDS (C-layout -> A-layout); PV = 8 WMMAs into 8 f32 accumulators.
// ---------------------------------------------------------------------------
__global__ __launch_bounds__(32) void attn_wmma(const __bf16* __restrict__ Qp,
                                                const __bf16* __restrict__ Kp,
                                                const __bf16* __restrict__ Vpt,
                                                __bf16* __restrict__ Obf,
                                                int B, int S, int H, int KV) {
  constexpr int hd = 128;
  const int lane = threadIdx.x & 31;
  const int q0 = blockIdx.x * 16;
  const int h  = blockIdx.y;
  const int b  = blockIdx.z;
  const int kv = h / (H / KV);

  const __bf16* Qbh = Qp  + (((size_t)b * H  + h)  * S) * hd;
  const __bf16* Kbh = Kp  + (((size_t)b * KV + kv) * S) * hd;
  const __bf16* Vbh = Vpt + (((size_t)b * KV + kv) * hd) * S;

  __shared__ __bf16 Kt[2][32 * hd];   // 32 L-rows x 128 (ld = 128)
  __shared__ __bf16 Vt[2][hd * 32];   // 128 d-rows x 32  (ld = 32)
  __shared__ __bf16 Pl[16 * 32];

  auto load_chunk = [&](int l0, int bufi) {
#if MHA_HAVE_TDM
    tdm_load_2d(lds_off(&Kt[bufi][0]), Kbh + (size_t)l0 * hd, hd, 32, hd);
    tdm_load_2d(lds_off(&Vt[bufi][0]), Vbh + l0, 32, hd, (unsigned)S);
#else
    for (int idx = lane; idx < 32 * hd; idx += 32)
      Kt[bufi][idx] = Kbh[(size_t)l0 * hd + idx];
    for (int idx = lane; idx < 32 * hd; idx += 32) {
      int d = idx >> 5, l = idx & 31;
      Vt[bufi][idx] = Vbh[(size_t)d * S + l0 + l];
    }
#endif
  };

  bf16x16 qf[4];
#pragma unroll
  for (int ks = 0; ks < 4; ++ks)
    qf[ks] = load_frag(Qbh + (size_t)q0 * hd + ks * 32, hd);

  f32x8 o[8];
#pragma unroll
  for (int t = 0; t < 8; ++t) o[t] = zero8();
  float mrow[8], lrow[8];
#pragma unroll
  for (int j = 0; j < 8; ++j) { mrow[j] = -__builtin_inff(); lrow[j] = 0.f; }

  const float scale = 0.08838834764831845f; // 1/sqrt(128)
  const int half = lane >> 4, c = lane & 15;

  load_chunk(0, 0);
  int buf = 0;
  for (int l0 = 0; l0 < S; l0 += 32) {
    if (l0 + 32 < S) {
      load_chunk(l0 + 32, buf ^ 1);
#if MHA_HAVE_TDM
      __builtin_amdgcn_s_wait_tensorcnt(2);   // current pair complete
#endif
    } else {
#if MHA_HAVE_TDM
      __builtin_amdgcn_s_wait_tensorcnt(0);
#endif
    }
    const __bf16* Kl = &Kt[buf][0];
    const __bf16* Vl = &Vt[buf][0];

    f32x8 s0 = zero8(), s1 = zero8();
#pragma unroll
    for (int ks = 0; ks < 4; ++ks) {
      bf16x16 b0 = load_frag(Kl + ks * 32, hd);
      bf16x16 b1 = load_frag(Kl + 16 * hd + ks * 32, hd);
      s0 = wmma_bf16(qf[ks], b0, s0);
      s1 = wmma_bf16(qf[ks], b1, s1);
    }

#pragma unroll
    for (int j = 0; j < 8; ++j) {
      float v0 = s0[j] * scale, v1 = s1[j] * scale;
      float t = fmaxf(v0, v1);
      t = fmaxf(t, SWZ_XOR(t, 1));
      t = fmaxf(t, SWZ_XOR(t, 2));
      t = fmaxf(t, SWZ_XOR(t, 4));
      t = fmaxf(t, SWZ_XOR(t, 8));
      float mnew  = fmaxf(mrow[j], t);
      float alpha = __expf(mrow[j] - mnew);
      float p0 = __expf(v0 - mnew), p1 = __expf(v1 - mnew);
      float ts = p0 + p1;
      ts += SWZ_XOR(ts, 1);
      ts += SWZ_XOR(ts, 2);
      ts += SWZ_XOR(ts, 4);
      ts += SWZ_XOR(ts, 8);
      lrow[j] = lrow[j] * alpha + ts;
      mrow[j] = mnew;
      int row = j + 8 * half;
      Pl[row * 32 + c]      = f2bf(p0);
      Pl[row * 32 + 16 + c] = f2bf(p1);
#pragma unroll
      for (int nt = 0; nt < 8; ++nt) o[nt][j] *= alpha;
    }
    __syncthreads();

    bf16x16 pA = load_frag(&Pl[0], 32);
#pragma unroll
    for (int nt = 0; nt < 8; ++nt) {
      bf16x16 bv = load_frag(Vl + (size_t)(nt * 16) * 32, 32);
      o[nt] = wmma_bf16(pA, bv, o[nt]);
    }
    __syncthreads();
    buf ^= 1;
  }

  // normalize + write [B,S,H*hd]
#pragma unroll
  for (int nt = 0; nt < 8; ++nt) {
#pragma unroll
    for (int j = 0; j < 8; ++j) {
      int m = j + 8 * half;
      float val = o[nt][j] / lrow[j];
      size_t idx = ((size_t)(b * S + q0 + m) * (size_t)(H * hd))
                 + (size_t)h * hd + nt * 16 + c;
      Obf[idx] = f2bf(val);
    }
  }
}

// ---------------------------------------------------------------------------
// Launch
// ---------------------------------------------------------------------------
extern "C" void kernel_launch(void* const* d_in, const int* in_sizes, int n_in,
                              void* d_out, int out_size, void* d_ws, size_t ws_size,
                              hipStream_t stream) {
  (void)in_sizes; (void)n_in; (void)out_size; (void)ws_size;
  constexpr int B = 2, S = 2048, D = 2048, H = 16, KV = 4, hd = 128;
  constexpr int M = B * S;
  constexpr int Nq = H * hd;             // 2048
  constexpr int Nkv = KV * hd;           // 512

  const float* x  = (const float*)d_in[0];
  const float* wq = (const float*)d_in[1];
  const float* wk = (const float*)d_in[2];
  const float* wv = (const float*)d_in[3];
  const float* wo = (const float*)d_in[4];
  float* out = (float*)d_out;

  char* ws = (char*)d_ws;
  size_t off = 0;
  auto alloc = [&](size_t bytes) {
    void* p = ws + off;
    off = (off + bytes + 255) & ~(size_t)255;
    return p;
  };
  __bf16* xbf  = (__bf16*)alloc((size_t)M * D * 2);
  __bf16* wqT  = (__bf16*)alloc((size_t)D * Nq * 2);
  __bf16* wkT  = (__bf16*)alloc((size_t)D * Nkv * 2);
  __bf16* wvT  = (__bf16*)alloc((size_t)D * Nkv * 2);
  __bf16* woT  = (__bf16*)alloc((size_t)Nq * D * 2);
  __bf16* qbf  = (__bf16*)alloc((size_t)M * Nq * 2);
  __bf16* kbf  = (__bf16*)alloc((size_t)M * Nkv * 2);
  __bf16* vbf  = (__bf16*)alloc((size_t)M * Nkv * 2);
  __bf16* Qp   = (__bf16*)alloc((size_t)M * Nq * 2);
  __bf16* Kp   = (__bf16*)alloc((size_t)M * Nkv * 2);
  __bf16* Vpt  = (__bf16*)alloc((size_t)M * Nkv * 2);
  __bf16* abf  = (__bf16*)alloc((size_t)M * Nq * 2);

  const int TB = 256;
  cvt_bf16<<<(M * D + TB - 1) / TB, TB, 0, stream>>>(x, xbf, M * D);
  cvtT_bf16<<<(D * Nq + TB - 1) / TB, TB, 0, stream>>>(wq, wqT, D, Nq);
  cvtT_bf16<<<(D * Nkv + TB - 1) / TB, TB, 0, stream>>>(wk, wkT, D, Nkv);
  cvtT_bf16<<<(D * Nkv + TB - 1) / TB, TB, 0, stream>>>(wv, wvT, D, Nkv);
  cvtT_bf16<<<(Nq * D + TB - 1) / TB, TB, 0, stream>>>(wo, woT, Nq, D);

  gemm_wmma<true><<<dim3(Nq / 64, M / 128), TB, 0, stream>>>(xbf, wqT, qbf, M, Nq, D);
  gemm_wmma<true><<<dim3(Nkv / 64, M / 128), TB, 0, stream>>>(xbf, wkT, kbf, M, Nkv, D);
  gemm_wmma<true><<<dim3(Nkv / 64, M / 128), TB, 0, stream>>>(xbf, wvT, vbf, M, Nkv, D);

  rope_pack<<<(B * S * H * (hd / 2) + TB - 1) / TB, TB, 0, stream>>>(qbf, Qp, B, S, H, hd);
  rope_pack<<<(B * S * KV * (hd / 2) + TB - 1) / TB, TB, 0, stream>>>(kbf, Kp, B, S, KV, hd);
  pack_vT<<<(B * S * KV * hd + TB - 1) / TB, TB, 0, stream>>>(vbf, Vpt, B, S, KV, hd);

  attn_wmma<<<dim3(S / 16, H, B), 32, 0, stream>>>(Qp, Kp, Vpt, abf, B, S, H, KV);

  gemm_wmma<false><<<dim3(D / 64, M / 128), TB, 0, stream>>>(abf, woT, out, M, D, Nq);
}
